// PureSparseLayer_58634893525466
// MI455X (gfx1250) — matrix operand order repaired and checked by reading, not compiled, not run on hardware
//
#include <hip/hip_runtime.h>
#include <stdint.h>

// ---------------------------------------------------------------------------
// Dense-WMMA formulation of the sparse layer:
//   out[32768,4096] = bf16(x)[32768,1024] @ bf16(W_dense)[1024,4096] + bias
// W_dense materialized (transposed, [unit][feature]) from (rows, kernel).
// Macro tile 128(M,batch) x 256(N,units), K chunks of 64 via TDM double
// buffering; 8 waves, each owning a 64x64 C tile (4x4 WMMA fragments) so the
// ds_load_b128 : v_wmma ratio is 1.0 instead of 1.5.
// ---------------------------------------------------------------------------

typedef unsigned int u32x4   __attribute__((ext_vector_type(4)));
typedef int          i32x4   __attribute__((ext_vector_type(4)));
typedef int          i32x8   __attribute__((ext_vector_type(8)));
typedef __bf16       bf16x16 __attribute__((ext_vector_type(16)));
typedef float        f32x8   __attribute__((ext_vector_type(8)));

#define BATCH       32768
#define INF         1024
#define UNITS       4096
#define ALPHA       32

#define TILE_M      128          // macro tile rows (batch)
#define TILE_N      256          // macro tile cols (units)
#define TILE_K      64           // K chunk per TDM load
#define ROW_BYTES   144          // 64 bf16 = 128B + 16B TDM pad -> conflict-free
#define A_BYTES     (TILE_M * ROW_BYTES)    // 18432 B
#define B_BYTES     (TILE_N * ROW_BYTES)    // 36864 B
#define NKCHUNK     (INF / TILE_K)          // 16

// ---- f32 -> bf16 round-to-nearest-even --------------------------------------
__device__ __forceinline__ uint16_t f2bf(float f) {
  uint32_t u = __float_as_uint(f);
  return (uint16_t)((u + 0x7fffu + ((u >> 16) & 1u)) >> 16);
}

// ---- Tensor Data Mover: 2D tile (TILE_K x tile_d1) global -> LDS ------------
// D# group0/group1 per CDNA5 ISA §8.3/8.4. data_size=2B, pad 4 dwords (16B)
// after every 32 dwords (128B = one 64-elem bf16 row) -> LDS row stride 144B.
__device__ __forceinline__ void tdm_load_2d(uint64_t gaddr, uint32_t lds_off,
                                            uint32_t tensor_d1 /*tensor rows*/,
                                            uint32_t tile_d1   /*tile rows*/) {
  const uint32_t td0 = 1024u;   // tensor fastest dim (features), elements
  u32x4 g0;
  g0[0] = 1u;                                    // count=1, user descriptor
  g0[1] = lds_off;                               // LDS byte address
  g0[2] = (uint32_t)gaddr;                       // global_addr[31:0]
  g0[3] = ((uint32_t)(gaddr >> 32) & 0x01FFFFFFu) | (2u << 30); // addr[56:32], type=2

  uint32_t w0 = (1u << 16)      // data_size = 2 bytes
              | (1u << 20)      // pad_enable
              | (4u << 22)      // pad_interval: 32 dwords (128B)
              | (3u << 25);     // pad_amount: 4 dwords (16B)
  uint32_t w1 = (td0 & 0xFFFFu) << 16;                        // tensor_dim0 lo16
  uint32_t w2 = (td0 >> 16) | ((tensor_d1 & 0xFFFFu) << 16);  // dim0 hi / dim1 lo
  uint32_t w3 = (tensor_d1 >> 16) | ((uint32_t)TILE_K << 16); // dim1 hi / tile_dim0
  uint32_t w4 = tile_d1;                                      // tile_dim1, tile_dim2=0
  uint32_t w5 = 1024u;                                        // tensor_dim0_stride lo32
  i32x8 g1;
  g1[0] = (int)w0; g1[1] = (int)w1; g1[2] = (int)w2; g1[3] = (int)w3;
  g1[4] = (int)w4; g1[5] = (int)w5; g1[6] = 0; g1[7] = 0;

  i32x4 z4 = {0, 0, 0, 0};
  i32x8 z8 = {0, 0, 0, 0, 0, 0, 0, 0};
  // clang-23 / therock-10.0 form: (g0, g1, g2, g3, g4, cpol)
  __builtin_amdgcn_tensor_load_to_lds(g0, g1, z4, z4, z8, 0);
}

// ---- prep kernels -----------------------------------------------------------
__global__ void zero_wt_kernel(uint4* __restrict__ p, int n) {
  int i = blockIdx.x * blockDim.x + threadIdx.x;
  if (i < n) { uint4 z = {0u, 0u, 0u, 0u}; p[i] = z; }
}

__global__ void scatter_w_kernel(const float* __restrict__ kern,
                                 const int* __restrict__ rows,
                                 uint16_t* __restrict__ wt) {
  int i = blockIdx.x * blockDim.x + threadIdx.x;   // 0 .. UNITS*ALPHA-1
  int u = i >> 5;
  int f = rows[i];
  wt[(size_t)u * INF + f] = f2bf(kern[i]);
}

__global__ void cvt_x_kernel(const float4* __restrict__ x,
                             uint64_t* __restrict__ xb, int n4) {
  int i = blockIdx.x * blockDim.x + threadIdx.x;
  if (i >= n4) return;
  float4 v = x[i];
  uint64_t pk = (uint64_t)f2bf(v.x) | ((uint64_t)f2bf(v.y) << 16) |
                ((uint64_t)f2bf(v.z) << 32) | ((uint64_t)f2bf(v.w) << 48);
  xb[i] = pk;
}

// ---- main GEMM: TDM double-buffered, bf16 WMMA, f32 accumulate --------------
__global__ __launch_bounds__(256) void
gemm_bf16_wmma_kernel(const uint16_t* __restrict__ xb,   // [32768,1024] bf16
                      const uint16_t* __restrict__ wt,   // [4096,1024]  bf16 (W^T)
                      const float* __restrict__ bias,    // [4096]
                      float* __restrict__ out) {         // [32768,4096] f32
  // A0 A1 B0 B1
  __shared__ __align__(16) unsigned char lds_pool[2 * A_BYTES + 2 * B_BYTES];
  const uint32_t ldsbase = (uint32_t)(uintptr_t)lds_pool;
  const uint32_t Boff = 2 * A_BYTES;

  const int tid   = threadIdx.x;
  const int lane  = tid & 31;
  const int wid   = tid >> 5;       // 8 waves
  const int waveM = wid >> 2;       // 0..1 -> 64-row slab
  const int waveN = wid & 3;        // 0..3 -> 64-col slab

  const int row0 = blockIdx.y * TILE_M;
  const int col0 = blockIdx.x * TILE_N;

  const int l16  = lane & 15;
  const int lh   = lane >> 4;       // 0/1 half-wave
  const int kA8  = lh * 8;          // A frag K base (16-bit A 16x32 layout)
  const int kB16 = lh * 16;         // unused; B frags use kA8, same K order as A

  f32x8 acc[4][4];
  const f32x8 z8f = {0.f, 0.f, 0.f, 0.f, 0.f, 0.f, 0.f, 0.f};
#pragma unroll
  for (int mt = 0; mt < 4; ++mt)
#pragma unroll
    for (int nt = 0; nt < 4; ++nt) acc[mt][nt] = z8f;

  // prologue: chunk 0 into buffers A0/B0 (one wave issues both TDM ops)
  if (wid == 0) {
    uint64_t ga = (uint64_t)(uintptr_t)xb + 2ull * ((uint64_t)row0 * INF);
    uint64_t gb = (uint64_t)(uintptr_t)wt + 2ull * ((uint64_t)col0 * INF);
    tdm_load_2d(ga, ldsbase, BATCH, TILE_M);
    tdm_load_2d(gb, ldsbase + Boff, UNITS, TILE_N);
  }

  for (int kc = 0; kc < NKCHUNK; ++kc) {
    const int cur = kc & 1;
    if (wid == 0) {
      if (kc + 1 < NKCHUNK) {
        const int nb = (kc + 1) & 1;
        uint64_t ga = (uint64_t)(uintptr_t)xb +
                      2ull * ((uint64_t)row0 * INF + (uint64_t)(kc + 1) * TILE_K);
        uint64_t gb = (uint64_t)(uintptr_t)wt +
                      2ull * ((uint64_t)col0 * INF + (uint64_t)(kc + 1) * TILE_K);
        tdm_load_2d(ga, ldsbase + nb * A_BYTES, BATCH, TILE_M);
        tdm_load_2d(gb, ldsbase + Boff + nb * B_BYTES, UNITS, TILE_N);
        __builtin_amdgcn_s_wait_tensorcnt(2);   // chunk kc's 2 ops complete
      } else {
        __builtin_amdgcn_s_wait_tensorcnt(0);
      }
    }
    __syncthreads();   // data for chunk kc visible to all waves

    const unsigned char* Ab = lds_pool + cur * A_BYTES;
    const unsigned char* Bb = lds_pool + Boff + cur * B_BYTES;

#pragma unroll
    for (int ks = 0; ks < 2; ++ks) {           // two K=32 steps per chunk
      const int koff = ks * 32;

      union BF { uint4 q[2]; bf16x16 v; };
      BF bfr[4];
#pragma unroll
      for (int nt = 0; nt < 4; ++nt) {         // B frags: col n=l16, K split 0-7/16-23
        const int c = waveN * 64 + nt * 16 + l16;
        const uint4* p = (const uint4*)(Bb + c * ROW_BYTES + (koff + kA8) * 2);
        bfr[nt].q[0] = p[0];
        bfr[nt].q[1] = p[2];
      }
#pragma unroll
      for (int mt = 0; mt < 4; ++mt) {         // A frags: row m=l16, K split 0-7/16-23
        const int r = waveM * 64 + mt * 16 + l16;
        const uint4* p = (const uint4*)(Ab + r * ROW_BYTES + (koff + kA8) * 2);
        BF af;
        af.q[0] = p[0];
        af.q[1] = p[2];                        // +32B -> K = kA8+16..kA8+23
#pragma unroll
        for (int nt = 0; nt < 4; ++nt) {
          acc[mt][nt] = __builtin_amdgcn_wmma_f32_16x16x32_bf16(
              false, af.v, false, bfr[nt].v, (short)0, acc[mt][nt], false, false);
        }
      }
    }
    __syncthreads();   // all waves done with buffer `cur` before it is reloaded
  }

  // epilogue: C layout (lanes0-15: M=r, lanes16-31: M=r+8), add bias, store f32
#pragma unroll
  for (int nt = 0; nt < 4; ++nt) {
    const int col = col0 + waveN * 64 + nt * 16 + l16;
    const float bv = bias[col];
#pragma unroll
    for (int mt = 0; mt < 4; ++mt) {
      const int mbase = row0 + waveM * 64 + mt * 16 + lh * 8;
#pragma unroll
      for (int r = 0; r < 8; ++r) {
        out[(size_t)(mbase + r) * UNITS + col] = acc[mt][nt][r] + bv;
      }
    }
  }
}

// ---------------------------------------------------------------------------
extern "C" void kernel_launch(void* const* d_in, const int* in_sizes, int n_in,
                              void* d_out, int out_size, void* d_ws, size_t ws_size,
                              hipStream_t stream) {
  const float* x      = (const float*)d_in[0];   // [32768,1024]
  const float* kernel = (const float*)d_in[1];   // [4096,32]
  const float* bias   = (const float*)d_in[2];   // [4096]
  const int*   rows   = (const int*)d_in[3];     // [4096,32]
  float* out = (float*)d_out;

  // workspace: x_bf16 (64MB) then W^T bf16 (8MB)
  uint16_t* xb = (uint16_t*)d_ws;
  uint16_t* wt = (uint16_t*)((char*)d_ws + (size_t)BATCH * INF * 2);

  // 1) zero dense W^T (8MB as uint4)
  {
    int n = (UNITS * INF * 2) / 16;   // 524288
    zero_wt_kernel<<<n / 256, 256, 0, stream>>>((uint4*)wt, n);
  }
  // 2) scatter sparse kernel values into W^T
  scatter_w_kernel<<<(UNITS * ALPHA) / 256, 256, 0, stream>>>(kernel, rows, wt);
  // 3) convert x to bf16
  {
    int n4 = (BATCH * INF) / 4;       // 8388608
    cvt_x_kernel<<<n4 / 256, 256, 0, stream>>>((const float4*)x, (uint64_t*)xb, n4);
  }
  // 4) GEMM with TDM + WMMA
  dim3 grid(UNITS / TILE_N, BATCH / TILE_M);   // (16, 256)
  gemm_bf16_wmma_kernel<<<grid, 256, 0, stream>>>(xb, wt, bias, out);
}
